// MoE_80204219286212
// MI455X (gfx1250) — matrix operand order repaired and checked by reading, not compiled, not run on hardware
//
#include <hip/hip_runtime.h>
#include <hip/hip_bf16.h>

// ---------------------------------------------------------------------------
// MoE (top-2 of 8 experts, D=H=1024, T=16384) for gfx1250 / MI455X.
// Dispatched expert execution (4x less compute than dense reference).
// GEMMs: bf16 v_wmma_f32_16x16x32_bf16, f32 accumulate.
// Weights pre-transposed+converted to bf16 once -> all LDS staging is b128.
// gfx1250 async-to-LDS staging (ASYNCcnt) when the builtin exists.
// ---------------------------------------------------------------------------

typedef __bf16 bf16_t;
typedef __attribute__((ext_vector_type(16))) __bf16 v16bf;
typedef __attribute__((ext_vector_type(8)))  __bf16 v8bf;
typedef __attribute__((ext_vector_type(8)))  float  v8f;
typedef __attribute__((ext_vector_type(4)))  int    v4i;

#define D_DIM 1024
#define H_DIM 1024
#define E_NUM 8
#define K_TOP 2
#define T_TOK 16384              // B*N = 4*4096
#define R_ROWS (T_TOK * K_TOP)   // 32768 compact expert-rows
#define LN_EPS 1e-5f
#define AT_LD 40                 // padded LDS row (80B): conflict-free frags

#if __has_builtin(__builtin_amdgcn_global_load_async_to_lds_b128)
#define MOE_ASYNC 1
#endif

typedef __attribute__((address_space(1))) v4i g_v4i;
typedef __attribute__((address_space(3))) v4i l_v4i;

// 16-byte LDS<-global copy (async if available, else through VGPRs)
__device__ __forceinline__ void cp16(bf16_t* dst_lds, const bf16_t* src) {
#ifdef MOE_ASYNC
  __builtin_amdgcn_global_load_async_to_lds_b128(
      (g_v4i*)src, (l_v4i*)dst_lds, 0, 0);
#else
  *reinterpret_cast<v8bf*>(dst_lds) = *reinterpret_cast<const v8bf*>(src);
#endif
}
__device__ __forceinline__ void cp_wait() {
#ifdef MOE_ASYNC
#if __has_builtin(__builtin_amdgcn_s_wait_asynccnt)
  __builtin_amdgcn_s_wait_asynccnt(0);
#else
  asm volatile("s_wait_asynccnt 0x0" ::: "memory");
#endif
#endif
}

// ---------------------------------------------------------------------------
// prep: x (f32) -> xb (bf16)
// ---------------------------------------------------------------------------
__global__ __launch_bounds__(256) void moe_cvt_x(const float* __restrict__ x,
                                                 bf16_t* __restrict__ xb) {
  size_t i = ((size_t)blockIdx.x * 256 + threadIdx.x) * 8;
  v8bf o;
#pragma unroll
  for (int j = 0; j < 8; ++j) o[j] = (bf16_t)x[i + j];
  *reinterpret_cast<v8bf*>(xb + i) = o;
}

// ---------------------------------------------------------------------------
// prep: W[e][k][n] (f32) -> Wt[e][n][k] (bf16), 32x32 tiles through LDS
// ---------------------------------------------------------------------------
__global__ __launch_bounds__(256) void moe_transpose_cvt(
    const float* __restrict__ W, bf16_t* __restrict__ Wt) {
  const int e  = blockIdx.z;
  const int n0 = blockIdx.x * 32;
  const int k0 = blockIdx.y * 32;
  __shared__ bf16_t tile[32][33];
  const int tid = threadIdx.x;

  int kk = tid >> 3;
  int nn = (tid & 7) * 4;
  const float* src = W + ((size_t)e * 1024 + k0 + kk) * 1024 + n0 + nn;
#pragma unroll
  for (int j = 0; j < 4; ++j) tile[kk][nn + j] = (bf16_t)src[j];
  __syncthreads();

  int n2 = tid >> 3;
  int k2 = (tid & 7) * 4;
  bf16_t* dst = Wt + ((size_t)e * 1024 + n0 + n2) * 1024 + k0 + k2;
#pragma unroll
  for (int j = 0; j < 4; ++j) dst[j] = tile[k2 + j][n2];
}

// ---------------------------------------------------------------------------
// 0) init counters
// ---------------------------------------------------------------------------
__global__ void moe_init(int* counts, int* cursors) {
  int i = threadIdx.x;
  if (i < E_NUM) { counts[i] = 0; cursors[i] = 0; }
}

// ---------------------------------------------------------------------------
// 1) gating: scores = x @ Wg + bg ; top-2 ; softmax weights ; expert counts
// ---------------------------------------------------------------------------
__global__ __launch_bounds__(128) void moe_gate(
    const float* __restrict__ x, const float* __restrict__ Wg,
    const float* __restrict__ bg,
    int* __restrict__ topi, float* __restrict__ topw, int* __restrict__ counts)
{
  const int t = blockIdx.x;
  const int tid = threadIdx.x;
  __shared__ float part[128][E_NUM];

  float acc[E_NUM];
#pragma unroll
  for (int e = 0; e < E_NUM; ++e) acc[e] = 0.f;

  for (int d = tid; d < D_DIM; d += 128) {
    float xv = x[(size_t)t * D_DIM + d];
#pragma unroll
    for (int e = 0; e < E_NUM; ++e) acc[e] += xv * Wg[d * E_NUM + e];
  }
#pragma unroll
  for (int e = 0; e < E_NUM; ++e) part[tid][e] = acc[e];
  __syncthreads();
  for (int s = 64; s > 0; s >>= 1) {
    if (tid < s) {
#pragma unroll
      for (int e = 0; e < E_NUM; ++e) part[tid][e] += part[tid + s][e];
    }
    __syncthreads();
  }
  if (tid == 0) {
    float sc[E_NUM];
#pragma unroll
    for (int e = 0; e < E_NUM; ++e) sc[e] = part[0][e] + bg[e];
    int i0 = 0;
#pragma unroll
    for (int e = 1; e < E_NUM; ++e) if (sc[e] > sc[i0]) i0 = e;
    int i1 = (i0 == 0) ? 1 : 0;
#pragma unroll
    for (int e = 0; e < E_NUM; ++e)
      if (e != i0 && sc[e] > sc[i1]) i1 = e;
    float m = sc[i0];
    float e1 = __expf(sc[i1] - m);
    float inv = 1.0f / (1.0f + e1);
    topi[t * 2 + 0] = i0; topi[t * 2 + 1] = i1;
    topw[t * 2 + 0] = inv; topw[t * 2 + 1] = e1 * inv;
    atomicAdd(&counts[i0], 1);
    atomicAdd(&counts[i1], 1);
  }
}

// ---------------------------------------------------------------------------
// 2) prefix sum over E=8 counts
// ---------------------------------------------------------------------------
__global__ void moe_prefix(const int* __restrict__ counts,
                           int* __restrict__ offsets, int* __restrict__ cursors) {
  if (threadIdx.x == 0 && blockIdx.x == 0) {
    int acc = 0;
    for (int e = 0; e < E_NUM; ++e) {
      offsets[e] = acc; cursors[e] = acc; acc += counts[e];
    }
    offsets[E_NUM] = acc;   // == R_ROWS
  }
}

// ---------------------------------------------------------------------------
// 3) scatter tokens into compact per-expert row lists
// ---------------------------------------------------------------------------
__global__ void moe_scatter(const int* __restrict__ topi, int* __restrict__ cursors,
                            int* __restrict__ list, int* __restrict__ row_of,
                            int* __restrict__ row_expert) {
  int t = blockIdx.x * blockDim.x + threadIdx.x;
  if (t >= T_TOK) return;
#pragma unroll
  for (int k = 0; k < K_TOP; ++k) {
    int e = topi[t * 2 + k];
    int pos = atomicAdd(&cursors[e], 1);
    list[pos] = t;
    row_of[t * 2 + k] = pos;
    row_expert[pos] = e;
  }
}

// ---------------------------------------------------------------------------
// WMMA grouped-GEMM core: 256 threads = 8 waves; block tile 128(M) x 128(N);
// wave tile 32x64 => 2x4 v8f accumulators; K stepped by 32 through LDS.
// A-src: bf16 rows (gathered via list if gather=true). B-src: Wt[e][n][k] bf16.
// ---------------------------------------------------------------------------
union ABFrag { v16bf v; v8bf h[2]; };

template <bool GATHER>
__device__ __forceinline__ void moe_gemm_core(
    const bf16_t* __restrict__ Asrc, const bf16_t* __restrict__ Wt,
    const float* __restrict__ bias,
    const int* __restrict__ offsets, const int* __restrict__ counts,
    const int* __restrict__ list, bf16_t* __restrict__ dst)
{
  const int e = blockIdx.z;
  const int cnt = counts[e];
  const int tile_m = blockIdx.y;
  if (tile_m * 128 >= cnt) return;
  const int row0  = offsets[e] + tile_m * 128;
  const int rlast = offsets[e] + cnt - 1;
  const int n0 = blockIdx.x * 128;

  __shared__ alignas(64) bf16_t At[128][AT_LD];
  __shared__ alignas(64) bf16_t Bt[128][AT_LD];
  __shared__ int arow[128];      // gathered source row per tile row

  const int tid = threadIdx.x;
  if (tid < 128) {
    int gr = row0 + tid;
    if (gr > rlast) gr = rlast;                 // clamp pad rows
    arow[tid] = GATHER ? list[gr] : gr;
  }
  __syncthreads();

  const int lane = tid & 31;
  const int wid  = tid >> 5;
  const int mwave = (wid & 3) * 32;
  const int nwave = (wid >> 2) * 64;

  v8f acc[2][4];
  const v8f vzero = {0.f, 0.f, 0.f, 0.f, 0.f, 0.f, 0.f, 0.f};
#pragma unroll
  for (int i = 0; i < 2; ++i)
#pragma unroll
    for (int j = 0; j < 4; ++j) acc[i][j] = vzero;

  const int r  = tid >> 1;            // staging row (0..127)
  const int c0 = (tid & 1) * 16;      // staging col half (0 or 16)
  const bf16_t* aRow = Asrc + (size_t)arow[r] * D_DIM;
  const bf16_t* bRow = Wt + ((size_t)e * 1024 + n0 + r) * 1024;

  for (int k0 = 0; k0 < D_DIM; k0 += 32) {
    // stage A (bf16 rows, contiguous 2x16B)
    cp16(&At[r][c0],     aRow + k0 + c0);
    cp16(&At[r][c0 + 8], aRow + k0 + c0 + 8);
    // stage B (pre-transposed bf16 weights, contiguous 2x16B)
    if (k0 + 32 < D_DIM) __builtin_prefetch(bRow + k0 + 32 + c0, 0, 0);
    cp16(&Bt[r][c0],     bRow + k0 + c0);
    cp16(&Bt[r][c0 + 8], bRow + k0 + c0 + 8);
    cp_wait();
    __syncthreads();

    v16bf afrag[2], bfrag[4];
    const int koff = (lane >> 4) * 8;    // A: lanes 16-31 hold K+8 halves
    const int kb   = (lane >> 4) * 8;    // B: lanes 16-31 hold K+8 halves
#pragma unroll
    for (int mt = 0; mt < 2; ++mt) {
      int M = mwave + mt * 16 + (lane & 15);
      ABFrag u;
      u.h[0] = *reinterpret_cast<const v8bf*>(&At[M][koff]);
      u.h[1] = *reinterpret_cast<const v8bf*>(&At[M][16 + koff]);
      afrag[mt] = u.v;
    }
#pragma unroll
    for (int nt = 0; nt < 4; ++nt) {
      int N = nwave + nt * 16 + (lane & 15);
      ABFrag u;
      u.h[0] = *reinterpret_cast<const v8bf*>(&Bt[N][kb]);
      u.h[1] = *reinterpret_cast<const v8bf*>(&Bt[N][16 + kb]);
      bfrag[nt] = u.v;
    }
#pragma unroll
    for (int mt = 0; mt < 2; ++mt)
#pragma unroll
      for (int nt = 0; nt < 4; ++nt)
        acc[mt][nt] = __builtin_amdgcn_wmma_f32_16x16x32_bf16(
            false, afrag[mt], false, bfrag[nt], (short)0, acc[mt][nt],
            false, false);
    __syncthreads();
  }

  // epilogue: + bias, store bf16 (C/D layout: VGPR i -> M = i + 8*(lane>=16))
  const int rhi = (lane >> 4) * 8;
  const int cl  = lane & 15;
#pragma unroll
  for (int mt = 0; mt < 2; ++mt)
#pragma unroll
    for (int nt = 0; nt < 4; ++nt)
#pragma unroll
      for (int i = 0; i < 8; ++i) {
        int lrow = mwave + mt * 16 + rhi + i;
        if (tile_m * 128 + lrow < cnt) {
          int col = n0 + nwave + nt * 16 + cl;
          float v = acc[mt][nt][i] + bias[e * 1024 + col];
          dst[(size_t)(row0 + lrow) * 1024 + col] = (bf16_t)v;
        }
      }
}

__global__ __launch_bounds__(256) void moe_gemm1(
    const bf16_t* __restrict__ xb, const bf16_t* __restrict__ Wt1,
    const float* __restrict__ b1,
    const int* __restrict__ offsets, const int* __restrict__ counts,
    const int* __restrict__ list, bf16_t* __restrict__ h_ws) {
  moe_gemm_core<true>(xb, Wt1, b1, offsets, counts, list, h_ws);
}

__global__ __launch_bounds__(256) void moe_gemm2(
    const bf16_t* __restrict__ h_ws, const bf16_t* __restrict__ Wt2,
    const float* __restrict__ b2,
    const int* __restrict__ offsets, const int* __restrict__ counts,
    bf16_t* __restrict__ y_ws) {
  moe_gemm_core<false>(h_ws, Wt2, b2, offsets, counts, nullptr, y_ws);
}

// ---------------------------------------------------------------------------
// 5) LayerNorm + ReLU over h rows (one 128-thread block per compact row)
// ---------------------------------------------------------------------------
__global__ __launch_bounds__(128) void moe_ln_relu(
    bf16_t* __restrict__ h_ws, const float* __restrict__ g1,
    const float* __restrict__ beta1, const int* __restrict__ row_expert)
{
  const int r = blockIdx.x;
  const int e = row_expert[r];
  const int tid = threadIdx.x;
  __shared__ float s1[128], s2[128];

  float v[8], sum = 0.f, sq = 0.f;
#pragma unroll
  for (int j = 0; j < 8; ++j) {
    float f = (float)h_ws[(size_t)r * H_DIM + tid + j * 128];
    v[j] = f; sum += f; sq += f * f;
  }
  s1[tid] = sum; s2[tid] = sq;
  __syncthreads();
  for (int s = 64; s > 0; s >>= 1) {
    if (tid < s) { s1[tid] += s1[tid + s]; s2[tid] += s2[tid + s]; }
    __syncthreads();
  }
  const float mean = s1[0] * (1.0f / H_DIM);
  const float var  = s2[0] * (1.0f / H_DIM) - mean * mean;
  const float rs   = rsqrtf(var + LN_EPS);
#pragma unroll
  for (int j = 0; j < 8; ++j) {
    int c = tid + j * 128;
    float y = (v[j] - mean) * rs * g1[e * H_DIM + c] + beta1[e * H_DIM + c];
    y = fmaxf(y, 0.f);
    h_ws[(size_t)r * H_DIM + c] = (bf16_t)y;
  }
}

// ---------------------------------------------------------------------------
// 7) combine: out[t] = x[t] + sum_k w_k * (LN(y[row(t,k)]) * g2 + beta2)
// ---------------------------------------------------------------------------
__global__ __launch_bounds__(128) void moe_combine(
    const float* __restrict__ x, const bf16_t* __restrict__ y_ws,
    const float* __restrict__ g2, const float* __restrict__ beta2,
    const int* __restrict__ topi, const float* __restrict__ topw,
    const int* __restrict__ row_of, float* __restrict__ out)
{
  const int t = blockIdx.x;
  const int tid = threadIdx.x;
  __shared__ float s1[128], s2[128];

  float o[8];
#pragma unroll
  for (int j = 0; j < 8; ++j) o[j] = x[(size_t)t * D_DIM + tid + j * 128];

  for (int k = 0; k < K_TOP; ++k) {
    const int e   = topi[t * 2 + k];
    const float w = topw[t * 2 + k];
    const int r   = row_of[t * 2 + k];

    float v[8], sum = 0.f, sq = 0.f;
#pragma unroll
    for (int j = 0; j < 8; ++j) {
      float f = (float)y_ws[(size_t)r * D_DIM + tid + j * 128];
      v[j] = f; sum += f; sq += f * f;
    }
    __syncthreads();                  // protect shared reuse across k
    s1[tid] = sum; s2[tid] = sq;
    __syncthreads();
    for (int s = 64; s > 0; s >>= 1) {
      if (tid < s) { s1[tid] += s1[tid + s]; s2[tid] += s2[tid + s]; }
      __syncthreads();
    }
    const float mean = s1[0] * (1.0f / D_DIM);
    const float var  = s2[0] * (1.0f / D_DIM) - mean * mean;
    const float rs   = rsqrtf(var + LN_EPS);
#pragma unroll
    for (int j = 0; j < 8; ++j) {
      int c = tid + j * 128;
      o[j] += w * ((v[j] - mean) * rs * g2[e * D_DIM + c] + beta2[e * D_DIM + c]);
    }
  }
#pragma unroll
  for (int j = 0; j < 8; ++j) out[(size_t)t * D_DIM + tid + j * 128] = o[j];
}

// ---------------------------------------------------------------------------
// launcher
// ---------------------------------------------------------------------------
extern "C" void kernel_launch(void* const* d_in, const int* in_sizes, int n_in,
                              void* d_out, int out_size, void* d_ws, size_t ws_size,
                              hipStream_t stream) {
  (void)in_sizes; (void)n_in; (void)out_size; (void)ws_size;
  const float* x     = (const float*)d_in[0];
  const float* Wg    = (const float*)d_in[1];
  const float* bg    = (const float*)d_in[2];
  const float* W1    = (const float*)d_in[3];
  const float* b1    = (const float*)d_in[4];
  const float* g1    = (const float*)d_in[5];
  const float* beta1 = (const float*)d_in[6];
  const float* W2    = (const float*)d_in[7];
  const float* b2    = (const float*)d_in[8];
  const float* g2    = (const float*)d_in[9];
  const float* beta2 = (const float*)d_in[10];
  float* out = (float*)d_out;

  char* ws = (char*)d_ws;
  size_t off = 0;
  auto carve = [&](size_t bytes) -> void* {
    off = (off + 255) & ~(size_t)255;
    void* p = ws + off;
    off += bytes;
    return p;
  };
  int*    counts     = (int*)carve(E_NUM * sizeof(int));
  int*    offsets    = (int*)carve((E_NUM + 1) * sizeof(int));
  int*    cursors    = (int*)carve(E_NUM * sizeof(int));
  int*    topi       = (int*)carve((size_t)T_TOK * K_TOP * sizeof(int));
  float*  topw       = (float*)carve((size_t)T_TOK * K_TOP * sizeof(float));
  int*    list       = (int*)carve((size_t)R_ROWS * sizeof(int));
  int*    row_of     = (int*)carve((size_t)R_ROWS * sizeof(int));
  int*    row_expert = (int*)carve((size_t)R_ROWS * sizeof(int));
  bf16_t* xb         = (bf16_t*)carve((size_t)T_TOK * D_DIM * sizeof(bf16_t));
  bf16_t* Wt1        = (bf16_t*)carve((size_t)E_NUM * D_DIM * H_DIM * sizeof(bf16_t));
  bf16_t* Wt2        = (bf16_t*)carve((size_t)E_NUM * H_DIM * D_DIM * sizeof(bf16_t));
  bf16_t* h_ws       = (bf16_t*)carve((size_t)R_ROWS * H_DIM * sizeof(bf16_t));
  bf16_t* y_ws       = (bf16_t*)carve((size_t)R_ROWS * D_DIM * sizeof(bf16_t));

  // prep passes (bandwidth-trivial, amortized over the GEMMs)
  moe_cvt_x<<<(T_TOK * D_DIM) / (256 * 8), 256, 0, stream>>>(x, xb);
  dim3 tgrid(1024 / 32, 1024 / 32, E_NUM);
  moe_transpose_cvt<<<tgrid, 256, 0, stream>>>(W1, Wt1);
  moe_transpose_cvt<<<tgrid, 256, 0, stream>>>(W2, Wt2);

  moe_init<<<1, 64, 0, stream>>>(counts, cursors);
  moe_gate<<<T_TOK, 128, 0, stream>>>(x, Wg, bg, topi, topw, counts);
  moe_prefix<<<1, 1, 0, stream>>>(counts, offsets, cursors);
  moe_scatter<<<(T_TOK + 255) / 256, 256, 0, stream>>>(topi, cursors, list,
                                                       row_of, row_expert);
  dim3 ggrid(1024 / 128, R_ROWS / 128, E_NUM);   // (8, 256, 8); early-exit blocks
  moe_gemm1<<<ggrid, 256, 0, stream>>>(xb, Wt1, b1, offsets, counts, list, h_ws);
  moe_ln_relu<<<R_ROWS, 128, 0, stream>>>(h_ws, g1, beta1, row_expert);
  moe_gemm2<<<ggrid, 256, 0, stream>>>(h_ws, Wt2, b2, offsets, counts, y_ws);
  moe_combine<<<T_TOK, 128, 0, stream>>>(x, y_ws, g2, beta2, topi, topw,
                                         row_of, out);
}
